// rate_RNN_mante_25099788878091
// MI455X (gfx1250) — matrix operand order repaired and checked by reading, not compiled, not run on hardware
//
#include <hip/hip_runtime.h>
#include <cmath>

// CDNA5 wave32 WMMA vector types
typedef __attribute__((ext_vector_type(16))) _Float16 v16h;
typedef __attribute__((ext_vector_type(8)))  _Float16 v8h;
typedef __attribute__((ext_vector_type(8)))  float    v8f;

namespace {
constexpr int T  = 512;
constexpr int NB = 64;     // batch
constexpr int IN = 4;
constexpr int H  = 2048;
constexpr int O  = 2;
constexpr int P  = 128;

constexpr int NWG = 16;            // persistent workgroups, one H-slice each
constexpr int HW  = H / NWG;       // 128 rows of H per WG
constexpr int NT  = 256;           // 8 wave32 per WG

constexpr float LM  = 0.904837418035959573f;  // exp(-dt/tau_m) = exp(-0.1)
constexpr float OML = 1.0f - LM;

// dynamic LDS: B1(pout colmajor) + B2(l*pin) f16 tiles, R, Z f16, small f32 tables
constexpr int SMEM_BYTES = (128 * 128 + 128 * 128 + NB * 128 + NB * 128) * 2
                         + (NB * IN + HW * IN + O * HW) * 4;   // = 102400 B
}

// ---- A fragment: 16-bit A 16x32. lane<16 -> K {0..15}; lane>=16 -> {16..31}
__device__ __forceinline__ v16h frag_a(const _Float16* p) {
  v8h lo = *(const v8h*)(p);
  v8h hi = *(const v8h*)(p + 8);
  v16h a;
#pragma unroll
  for (int i = 0; i < 8; ++i) { a[i] = lo[i]; a[i + 8] = hi[i]; }
  return a;
}

// ---- B fragment: 16-bit B 32x16. lane n=L&15, K = 16 contiguous (half selected by L>>4)
__device__ __forceinline__ v16h frag_b(const _Float16* p) {
  v8h lo = *(const v8h*)(p);
  v8h hi = *(const v8h*)(p + 8);
  v16h b;
#pragma unroll
  for (int i = 0; i < 8; ++i) { b[i] = lo[i]; b[i + 8] = hi[i]; }
  return b;
}

// device-wide sense-reversing barrier across the 16 persistent WGs
__device__ __forceinline__ void grid_barrier(unsigned* cnt, volatile unsigned* gen) {
  __syncthreads();
  if (threadIdx.x == 0) {
    __threadfence();
    unsigned g = *gen;
    if (atomicAdd(cnt, 1u) == NWG - 1) {
      *cnt = 0u;
      __threadfence();
      *gen = g + 1u;
    } else {
      while (*gen == g) { __builtin_amdgcn_s_sleep(1); }
    }
    __threadfence();
  }
  __syncthreads();
}

extern "C" __global__ void rnn_init(unsigned* bar) {
  bar[0] = 0u;
  bar[1] = 0u;
}

extern "C" __global__ __launch_bounds__(NT)
void rate_rnn_persistent(const float* __restrict__ inputs,  // [T][NB][IN]
                         const float* __restrict__ Win,     // [H][IN]
                         const float* __restrict__ Wout,    // [O][H]
                         const float* __restrict__ pin,     // [H][P]
                         const float* __restrict__ pout,    // [H][P]
                         const float* __restrict__ lvec,    // [P]
                         float* __restrict__ Zslots,        // [2][NWG][NB*P]
                         float* __restrict__ routp,         // [NWG][T][NB][O]
                         unsigned* __restrict__ bar) {
  extern __shared__ char smem[];
  _Float16* B1 = (_Float16*)smem;          // [n=P][k=HW] colmajor pout slice (GEMM1 B)
  _Float16* B2 = B1 + 128 * 128;           // [n=HW][k=P] (l*pin) slice       (GEMM2 B)
  _Float16* Rl = B2 + 128 * 128;           // [b=64][h=128] tanh(mem)         (GEMM1 A)
  _Float16* Zl = Rl + NB * 128;            // [b=64][p=128] reduced Z         (GEMM2 A)
  float* xl    = (float*)(Zl + NB * 128);  // [64][4]   inputs[t]
  float* winl  = xl + NB * IN;             // [128][4]  Win slice
  float* woutl = winl + HW * IN;           // [2][128]  Wout slice

  const int wg   = blockIdx.x;
  const int tid  = threadIdx.x;
  const int lane = tid & 31;
  const int wave = tid >> 5;
  const int mt   = wave & 3;     // 16-row batch tile owned by this wave
  const int ng   = wave >> 2;    // which 64-column half of the N dimension
  const int l15  = lane & 15;
  const int lhi  = (lane >> 4) & 1;
  const int h0   = wg * HW;

  // ---- stage resident weights into LDS (one time) ----
  for (int e = tid; e < 128 * 128; e += NT) {        // GEMM1 B: column-major pout slice
    int n = e & 127, k = e >> 7;
    B1[n * 128 + k] = (_Float16)pout[(h0 + k) * P + n];
  }
  for (int e = tid; e < 128 * 128; e += NT) {        // GEMM2 B: (l*pin) row-major
    int p = e & 127, h = e >> 7;
    B2[h * 128 + p] = (_Float16)(pin[(h0 + h) * P + p] * lvec[p]);
  }
  for (int e = tid; e < HW * IN; e += NT) winl[e] = Win[h0 * IN + e];
  for (int e = tid; e < O * HW; e += NT) {
    int o = e >> 7, j = e & 127;
    woutl[o * 128 + j] = Wout[o * H + h0 + j];
  }
  for (int e = tid; e < NB * 128; e += NT) Rl[e] = (_Float16)0.f;  // tanh(mem0) = 0
  __syncthreads();

  // mem in WMMA C/D layout: memv[nt][r] -> (b = mt*16 + lhi*8 + r, h = ng*64 + nt*16 + l15)
  float memv[4][8];
#pragma unroll
  for (int nt = 0; nt < 4; ++nt)
#pragma unroll
    for (int r = 0; r < 8; ++r) memv[nt][r] = 0.f;

  const int bbase = mt * 16 + lhi * 8;
  const int cbase = ng * 64 + l15;

  for (int t = 0; t < T; ++t) {
    const int par = t & 1;

    if (tid < NB * IN) xl[tid] = inputs[(size_t)t * NB * IN + tid];

    // ---- GEMM1: Zpart = tanh(mem_{t-1}) @ pout_slice  (K = local H) ----
    v16h a1[4];
#pragma unroll
    for (int kt = 0; kt < 4; ++kt)
      a1[kt] = frag_a(Rl + (mt * 16 + l15) * 128 + kt * 32 + lhi * 16);

    float* slot = Zslots + ((size_t)par * NWG + wg) * (NB * P);
#pragma unroll
    for (int nt = 0; nt < 4; ++nt) {
      v8f acc = {};
#pragma unroll
      for (int kt = 0; kt < 4; ++kt) {
        v16h bf = frag_b(B1 + (ng * 64 + nt * 16 + l15) * 128 + kt * 32 + lhi * 16);
        acc = __builtin_amdgcn_wmma_f32_16x16x32_f16(false, a1[kt], false, bf,
                                                     (short)0, acc, false, false);
      }
#pragma unroll
      for (int r = 0; r < 8; ++r)
        slot[(bbase + r) * P + cbase + nt * 16] = acc[r];
    }

    grid_barrier(bar, bar + 1);           // partial Z visible device-wide

    // ---- reduce rank-128 bottleneck Z over all WGs (fixed order -> deterministic) ----
    const float* zbase = Zslots + (size_t)par * NWG * (NB * P);
#pragma unroll
    for (int j = 0; j < (NB * P) / NT; ++j) {
      int e = tid + NT * j;
      float s = 0.f;
#pragma unroll
      for (int w = 0; w < NWG; ++w) s += zbase[w * (NB * P) + e];
      Zl[e] = (_Float16)s;
    }
    __syncthreads();

    // ---- GEMM2 + leaky integration: mem = lm*mem + (1-lm)*(Win@x + Z@(l*pin)^T) ----
    v16h a2[4];
#pragma unroll
    for (int kt = 0; kt < 4; ++kt)
      a2[kt] = frag_a(Zl + (mt * 16 + l15) * 128 + kt * 32 + lhi * 16);
#pragma unroll
    for (int nt = 0; nt < 4; ++nt) {
      v8f acc = {};
#pragma unroll
      for (int kt = 0; kt < 4; ++kt) {
        v16h bf = frag_b(B2 + (ng * 64 + nt * 16 + l15) * 128 + kt * 32 + lhi * 16);
        acc = __builtin_amdgcn_wmma_f32_16x16x32_f16(false, a2[kt], false, bf,
                                                     (short)0, acc, false, false);
      }
#pragma unroll
      for (int r = 0; r < 8; ++r) {
        int bi = bbase + r;
        int hl = cbase + nt * 16;
        float ix = 0.f;
#pragma unroll
        for (int i = 0; i < IN; ++i) ix = fmaf(xl[bi * IN + i], winl[hl * IN + i], ix);
        memv[nt][r] = LM * memv[nt][r] + OML * (ix + acc[r]);
      }
    }

    // ---- R = tanh(mem_t): feeds next step's GEMM1 and this step's readout ----
#pragma unroll
    for (int nt = 0; nt < 4; ++nt)
#pragma unroll
      for (int r = 0; r < 8; ++r)
        Rl[(bbase + r) * 128 + cbase + nt * 16] = (_Float16)tanhf(memv[nt][r]);
    __syncthreads();

    // ---- per-WG readout partial: rout_part = R_slice @ Wout_slice^T ----
    if (tid < NB * O) {
      int bi = tid >> 1, o = tid & 1;
      float s = 0.f;
      for (int h = 0; h < HW; ++h)
        s += (float)Rl[bi * 128 + h] * woutl[o * 128 + h];
      routp[(((size_t)wg * T + t) * NB + bi) * O + o] = s;
    }
  }
}

extern "C" __global__ __launch_bounds__(256)
void rnn_reduce_out(const float* __restrict__ routp, float* __restrict__ out) {
  int e = blockIdx.x * blockDim.x + threadIdx.x;   // over T*NB*O = 65536
  float s = 0.f;
#pragma unroll
  for (int w = 0; w < NWG; ++w) s += routp[(size_t)w * T * NB * O + e];
  out[e] = s;
}

extern "C" void kernel_launch(void* const* d_in, const int* in_sizes, int n_in,
                              void* d_out, int out_size, void* d_ws, size_t ws_size,
                              hipStream_t stream) {
  (void)in_sizes; (void)n_in; (void)out_size; (void)ws_size;
  const float* inputs = (const float*)d_in[0];
  const float* Win    = (const float*)d_in[1];
  const float* Wout   = (const float*)d_in[2];
  const float* pin    = (const float*)d_in[3];
  const float* pout   = (const float*)d_in[4];
  const float* lvec   = (const float*)d_in[5];
  float* out = (float*)d_out;

  float* Zslots = (float*)d_ws;                          // 2*NWG*NB*P f32   (1 MB)
  float* routp  = Zslots + 2 * NWG * NB * P;             // NWG*T*NB*O f32   (4 MB)
  unsigned* bar = (unsigned*)(routp + (size_t)NWG * T * NB * O);

  rnn_init<<<dim3(1), dim3(1), 0, stream>>>(bar);
  rate_rnn_persistent<<<dim3(NWG), dim3(NT), SMEM_BYTES, stream>>>(
      inputs, Win, Wout, pin, pout, lvec, Zslots, routp, bar);
  rnn_reduce_out<<<dim3((T * NB * O) / 256), dim3(256), 0, stream>>>(routp, out);
}
